// PointNetPlusPlus_85426899517944
// MI455X (gfx1250) — matrix operand order repaired and checked by reading, not compiled, not run on hardware
//
#include <hip/hip_runtime.h>
#include <hip/hip_bf16.h>

typedef __attribute__((ext_vector_type(16))) _Float16 v16h;
typedef __attribute__((ext_vector_type(8)))  _Float16 v8h;
typedef __attribute__((ext_vector_type(8)))  float    v8f;

#define BATCH 16
#define EPSBN 1e-5f

// ---------------------------------------------------------------------------
// Farthest point sampling: one block per batch. Distance array lives in LDS
// (4096 f32 = 16KB of 320KB/WGP). Emitted index is the *incoming* far
// (matches the jax.lax.scan carry semantics); argmax ties -> lowest index.
// ---------------------------------------------------------------------------
__global__ void k_fps(const float* __restrict__ xyz, int N, int npoint,
                      int* __restrict__ outIdx) {
  __shared__ float dist[4096];
  __shared__ float rval[256];
  __shared__ int   ridx[256];
  const int b = blockIdx.x, t = threadIdx.x;
  const float* base = xyz + (size_t)b * N * 3;
  for (int j = t; j < N; j += 256) dist[j] = 1e10f;
  __syncthreads();
  int far = 0;
  for (int i = 0; i < npoint; ++i) {
    if (t == 0) outIdx[b * npoint + i] = far;
    const float cx = base[far * 3 + 0];
    const float cy = base[far * 3 + 1];
    const float cz = base[far * 3 + 2];
    float best = -1.0f; int bidx = N - 1;
    for (int j = t; j < N; j += 256) {
      const float dx = base[j * 3 + 0] - cx;
      const float dy = base[j * 3 + 1] - cy;
      const float dz = base[j * 3 + 2] - cz;
      const float d  = dx * dx + dy * dy + dz * dz;
      const float nd = fminf(dist[j], d);
      dist[j] = nd;
      if (nd > best) { best = nd; bidx = j; }
    }
    rval[t] = best; ridx[t] = bidx;
    __syncthreads();
    for (int s = 128; s > 0; s >>= 1) {
      if (t < s) {
        if (rval[t + s] > rval[t] ||
            (rval[t + s] == rval[t] && ridx[t + s] < ridx[t])) {
          rval[t] = rval[t + s]; ridx[t] = ridx[t + s];
        }
      }
      __syncthreads();
    }
    far = ridx[0];
    __syncthreads();
  }
}

__global__ void k_gather_xyz(const float* __restrict__ xyz, const int* __restrict__ idx,
                             float* __restrict__ out, int N, int P, int BP) {
  int i = blockIdx.x * blockDim.x + threadIdx.x;
  if (i >= BP) return;
  const int b = i / P;
  const int id = idx[i];
  const float* s = xyz + ((size_t)b * N + id) * 3;
  out[(size_t)i * 3 + 0] = s[0];
  out[(size_t)i * 3 + 1] = s[1];
  out[(size_t)i * 3 + 2] = s[2];
}

// ---------------------------------------------------------------------------
// Ball query + gather + concat + K-pad, one wave32 per centroid. Ordered
// selection of the first S in-radius indices via ballot + prefix popcount
// (equivalent to the reference's where/sort/take-first-nsample). Output rows
// are f16 (GEMM input format). Feature source type templated (f32 raw input
// for SA1, f16 activations for SA2).
// ---------------------------------------------------------------------------
template <typename TF>
__global__ void k_group(const float* __restrict__ xyz, const TF* __restrict__ feats,
                        const float* __restrict__ nxyz, _Float16* __restrict__ out,
                        int Nsrc, int C, int P, int S, int Kp, float r2) {
  const int bp = blockIdx.x;            // b*P + p
  const int b = bp / P;
  const int lane = threadIdx.x;         // wave32
  const float* xb = xyz + (size_t)b * Nsrc * 3;
  const TF* fb = feats + (size_t)b * Nsrc * C;
  const float cx = nxyz[(size_t)bp * 3 + 0];
  const float cy = nxyz[(size_t)bp * 3 + 1];
  const float cz = nxyz[(size_t)bp * 3 + 2];
  _Float16* ob = out + (size_t)bp * S * Kp;

  int cnt = 0, first = -1;
  for (int base = 0; base < Nsrc; base += 32) {
    const int j = base + lane;
    bool in = false;
    float dx = 0.f, dy = 0.f, dz = 0.f;
    if (j < Nsrc) {
      dx = xb[j * 3 + 0] - cx;
      dy = xb[j * 3 + 1] - cy;
      dz = xb[j * 3 + 2] - cz;
      in = (dx * dx + dy * dy + dz * dz) <= r2;
    }
    const unsigned mask = (unsigned)__ballot(in);
    const int pre = __popc(mask & ((1u << lane) - 1u));
    const int pos = cnt + pre;
    if (in && pos < S) {
      _Float16* row = ob + (size_t)pos * Kp;
      row[0] = (_Float16)dx; row[1] = (_Float16)dy; row[2] = (_Float16)dz;
      for (int c = 0; c < C; ++c) row[3 + c] = (_Float16)(float)fb[(size_t)j * C + c];
      for (int c = 3 + C; c < Kp; ++c) row[c] = (_Float16)0.f;
    }
    if (first < 0 && mask) first = base + __builtin_ctz(mask);
    cnt += __popc(mask);
    if (cnt >= S) break;
  }
  if (cnt < S) {
    const int fj = (first < 0) ? 0 : first;
    const float dx = xb[fj * 3 + 0] - cx;
    const float dy = xb[fj * 3 + 1] - cy;
    const float dz = xb[fj * 3 + 2] - cz;
    for (int s = cnt + lane; s < S; s += 32) {
      _Float16* row = ob + (size_t)s * Kp;
      row[0] = (_Float16)dx; row[1] = (_Float16)dy; row[2] = (_Float16)dz;
      for (int c = 0; c < C; ++c) row[3 + c] = (_Float16)(float)fb[(size_t)fj * C + c];
      for (int c = 3 + C; c < Kp; ++c) row[c] = (_Float16)0.f;
    }
  }
}

// ---------------------------------------------------------------------------
// Weight prep: f32 W[K,N] -> f16 WT[Np,Kp] transposed + zero-padded, so GEMM
// B-fragment loads are contiguous b128s.
// ---------------------------------------------------------------------------
__global__ void k_prep_w(const float* __restrict__ W, _Float16* __restrict__ WT,
                         int K, int N, int Kp, int Np) {
  const int i = blockIdx.x * blockDim.x + threadIdx.x;
  if (i >= Kp * Np) return;
  const int n = i / Kp, k = i % Kp;
  WT[i] = (_Float16)((k < K && n < N) ? W[(size_t)k * N + n] : 0.f);
}

__global__ void k_pad_vec(const float* __restrict__ b, float* __restrict__ bp,
                          int N, int Np) {
  const int i = blockIdx.x * blockDim.x + threadIdx.x;
  if (i >= Np) return;
  bp[i] = (i < N) ? b[i] : 0.f;
}

// ---------------------------------------------------------------------------
// WMMA GEMM: Y[M,N](f32) = X[M,K](f16) @ WT[N,K](f16)^T + bias.
// One wave per 16x16 tile; K-loop step 32, unroll 2 (two v_wmma back-to-back).
// Fragment packing (A/B in contiguous k order, same for both; not the ISA table):
//   A 16x32 (f16): lane holds row M=lane%16; lanes<16 carry K 0..15,
//                  lanes>=16 carry K 16..31          -> two contiguous v8h.
//   B 32x16 (f16): lane holds col N=lane%16; lanes<16 carry K 0..15,
//                  lanes>=16 carry K 16..31          -> two contiguous v8h
//                  from the transposed WT row.
//   C/D: VGPR r, lanes<16 -> M=r, lanes>=16 -> M=r+8, N=lane%16.
// M multiple of 16, K multiple of 32, N multiple of 16 (workspace padding).
// ---------------------------------------------------------------------------
__global__ void k_gemm_wmma(const _Float16* __restrict__ X,
                            const _Float16* __restrict__ WT,
                            const float* __restrict__ bias, float* __restrict__ Y,
                            int M, int K, int Nout) {
  const int tm = blockIdx.x, tn = blockIdx.y;
  const int lane = threadIdx.x & 31;
  const int laneLo = lane & 15;
  const int laneHi = lane >> 4;
  const _Float16* xr = X + (size_t)(tm * 16 + laneLo) * K + (laneHi ? 16 : 0);
  const _Float16* wr = WT + (size_t)(tn * 16 + laneLo) * K + (laneHi ? 16 : 0);
  v8f acc = {};
#pragma unroll 2
  for (int k0 = 0; k0 < K; k0 += 32) {
    __builtin_prefetch(xr + k0 + 128, 0, 1);             // global_prefetch
    const v8h a0 = *(const v8h*)(xr + k0);
    const v8h a1 = *(const v8h*)(xr + k0 + 8);
    const v8h b0 = *(const v8h*)(wr + k0);
    const v8h b1 = *(const v8h*)(wr + k0 + 8);
    const v16h a = __builtin_shufflevector(a0, a1, 0, 1, 2, 3, 4, 5, 6, 7,
                                           8, 9, 10, 11, 12, 13, 14, 15);
    const v16h b = __builtin_shufflevector(b0, b1, 0, 1, 2, 3, 4, 5, 6, 7,
                                           8, 9, 10, 11, 12, 13, 14, 15);
    acc = __builtin_amdgcn_wmma_f32_16x16x32_f16(
        /*neg_a=*/false, a, /*neg_b=*/false, b,
        /*c_mod=*/(short)0, acc, /*reuse_a=*/false, /*reuse_b=*/false);
  }
  const int col = tn * 16 + laneLo;
  const float bv = bias[col];
#pragma unroll
  for (int r = 0; r < 8; ++r) {
    const int orow = tm * 16 + laneHi * 8 + r;
    Y[(size_t)orow * Nout + col] = acc[r] + bv;
  }
}

// ---------------------------------------------------------------------------
// BatchNorm over all rows (f32 stats on the raw GEMM output), then fused
// normalize + ReLU emitting the next layer's f16 input.
// ---------------------------------------------------------------------------
__global__ void k_zero(float* __restrict__ p, int n) {
  int i = blockIdx.x * blockDim.x + threadIdx.x;
  if (i < n) p[i] = 0.f;
}

__global__ void k_bn_stats(const float* __restrict__ Y, long M, int C,
                           float* __restrict__ stats) {
  __shared__ float ls[2048];                       // C <= 1024
  for (int c = threadIdx.x; c < 2 * C; c += blockDim.x) ls[c] = 0.f;
  __syncthreads();
  const long tot = M * (long)C;
  for (long i = (long)blockIdx.x * blockDim.x + threadIdx.x; i < tot;
       i += (long)gridDim.x * blockDim.x) {
    const float v = Y[i];
    const int c = (int)(i % C);
    atomicAdd(&ls[c], v);
    atomicAdd(&ls[C + c], v * v);
  }
  __syncthreads();
  for (int c = threadIdx.x; c < 2 * C; c += blockDim.x) atomicAdd(&stats[c], ls[c]);
}

__global__ void k_bn_relu(const float* __restrict__ Y, _Float16* __restrict__ Xo,
                          long tot, int C, const float* __restrict__ stats,
                          const float* __restrict__ g, const float* __restrict__ be,
                          float invM) {
  const long i = (long)blockIdx.x * blockDim.x + threadIdx.x;
  if (i >= tot) return;
  const int c = (int)(i % C);
  const float m = stats[c] * invM;
  const float var = stats[C + c] * invM - m * m;
  const float x = g[c] * (Y[i] - m) * rsqrtf(var + EPSBN) + be[c];
  Xo[i] = (_Float16)fmaxf(x, 0.f);
}

__global__ void k_maxpool_h(const _Float16* __restrict__ in, _Float16* __restrict__ out,
                            int BP, int S, int C) {
  const int i = blockIdx.x * blockDim.x + threadIdx.x;
  if (i >= BP * C) return;
  const int bp = i / C;
  const int c = i % C;
  float m = -3.4e38f;
  for (int s = 0; s < S; ++s)
    m = fmaxf(m, (float)in[((size_t)bp * S + s) * C + c]);
  out[i] = (_Float16)m;
}

__global__ void k_concat3(const float* __restrict__ xyz, const _Float16* __restrict__ feat,
                          _Float16* __restrict__ out, int rows, int C, int Kp) {
  const int i = blockIdx.x * blockDim.x + threadIdx.x;
  if (i >= rows * Kp) return;
  const int r = i / Kp, c = i % Kp;
  float v = 0.f;
  if (c < 3) v = xyz[(size_t)r * 3 + c];
  else if (c < 3 + C) v = (float)feat[(size_t)r * C + (c - 3)];
  out[i] = (_Float16)v;
}

__global__ void k_copy_out(const float* __restrict__ Yp, float* __restrict__ out,
                           int M, int N, int Np) {
  const int i = blockIdx.x * blockDim.x + threadIdx.x;
  if (i >= M * N) return;
  const int r = i / N, c = i % N;
  out[i] = Yp[(size_t)r * Np + c];
}

// ---------------------------------------------------------------------------
// Host orchestration. Input leaf order from setup_inputs():
//  0 xyz, 1 features, 2..13 sa1 (3x{W,b,g,be}), 14..25 sa2, 26..37 sa3,
//  38..41 fc1, 42..45 fc2, 46..47 fc3 {W,b}
// ---------------------------------------------------------------------------
extern "C" void kernel_launch(void* const* d_in, const int* in_sizes, int n_in,
                              void* d_out, int out_size, void* d_ws, size_t ws_size,
                              hipStream_t stream) {
  (void)in_sizes; (void)out_size; (void)ws_size;
  const float* Pm[48];
  for (int i = 0; i < n_in && i < 48; ++i) Pm[i] = (const float*)d_in[i];
  const float* xyz  = Pm[0];
  const float* feat = Pm[1];

  size_t off = 0;
  auto af = [&](size_t n) -> float* {          // f32 alloc
    float* p = (float*)((char*)d_ws + off);
    off = (off + n * 4 + 255) & ~(size_t)255;
    return p;
  };
  auto ah = [&](size_t n) -> _Float16* {       // f16 alloc
    _Float16* p = (_Float16*)((char*)d_ws + off);
    off = (off + n * 2 + 255) & ~(size_t)255;
    return p;
  };
  float*     YF   = af(33554432);              // f32 GEMM output (max 256k x 128)
  _Float16*  XA   = ah(33554432);              // f16 activation ping
  _Float16*  XB   = ah(33554432);              // f16 activation pong
  _Float16*  l1f  = ah(16 * 512 * 128);
  float*     l1x  = af(16 * 512 * 3);
  _Float16*  l2f  = ah(16 * 128 * 256);
  float*     l2x  = af(16 * 128 * 3);
  _Float16*  l3   = ah(16 * 1024);
  _Float16*  wt   = ah(1024 * 1024);           // transposed padded weight staging
  float*     bpad = af(64);
  float*     stats= af(2048);
  _Float16*  h1   = ah(16 * 512);
  _Float16*  h2   = ah(16 * 256);
  int*       idx1 = (int*)af(16 * 512);
  int*       idx2 = (int*)af(16 * 128);

  auto prepw = [&](const float* W, int K, int N, int Kp, int Np) {
    k_prep_w<<<(Kp * Np + 255) / 256, 256, 0, stream>>>(W, wt, K, N, Kp, Np);
  };
  auto gemm = [&](const _Float16* X, const float* b, float* Y, int M, int K, int N) {
    dim3 grid(M / 16, N / 16);
    k_gemm_wmma<<<grid, 32, 0, stream>>>(X, wt, b, Y, M, K, N);
  };
  auto bnrelu = [&](const float* Y, _Float16* Xo, long M, int C,
                    const float* g_, const float* be_) {
    k_zero<<<8, 256, 0, stream>>>(stats, 2048);
    k_bn_stats<<<256, 256, 0, stream>>>(Y, M, C, stats);
    const long tot = M * (long)C;
    k_bn_relu<<<(int)((tot + 255) / 256), 256, 0, stream>>>(Y, Xo, tot, C, stats,
                                                            g_, be_, 1.0f / (float)M);
  };

  // ---- SA1: npoint=512, r=0.2, S=32, C_in=3+61=64 (no K pad) ----
  k_fps<<<BATCH, 256, 0, stream>>>(xyz, 4096, 512, idx1);
  k_gather_xyz<<<(16 * 512 + 255) / 256, 256, 0, stream>>>(xyz, idx1, l1x, 4096, 512, 16 * 512);
  k_group<float><<<16 * 512, 32, 0, stream>>>(xyz, feat, l1x, XA, 4096, 61, 512, 32, 64, 0.04f);
  const long M1 = 16L * 512 * 32;
  prepw(Pm[2], 64, 64, 64, 64);    gemm(XA, Pm[3],  YF, (int)M1, 64, 64);
  bnrelu(YF, XB, M1, 64, Pm[4], Pm[5]);
  prepw(Pm[6], 64, 64, 64, 64);    gemm(XB, Pm[7],  YF, (int)M1, 64, 64);
  bnrelu(YF, XA, M1, 64, Pm[8], Pm[9]);
  prepw(Pm[10], 64, 128, 64, 128); gemm(XA, Pm[11], YF, (int)M1, 64, 128);
  bnrelu(YF, XB, M1, 128, Pm[12], Pm[13]);
  k_maxpool_h<<<(16 * 512 * 128 + 255) / 256, 256, 0, stream>>>(XB, l1f, 16 * 512, 32, 128);

  // ---- SA2: npoint=128, r=0.4, S=64, C_in=3+128=131 -> Kp=160 ----
  k_fps<<<BATCH, 256, 0, stream>>>(l1x, 512, 128, idx2);
  k_gather_xyz<<<(16 * 128 + 255) / 256, 256, 0, stream>>>(l1x, idx2, l2x, 512, 128, 16 * 128);
  k_group<_Float16><<<16 * 128, 32, 0, stream>>>(l1x, l1f, l2x, XA, 512, 128, 128, 64, 160, 0.16f);
  const long M2 = 16L * 128 * 64;
  prepw(Pm[14], 131, 128, 160, 128); gemm(XA, Pm[15], YF, (int)M2, 160, 128);
  bnrelu(YF, XB, M2, 128, Pm[16], Pm[17]);
  prepw(Pm[18], 128, 128, 128, 128); gemm(XB, Pm[19], YF, (int)M2, 128, 128);
  bnrelu(YF, XA, M2, 128, Pm[20], Pm[21]);
  prepw(Pm[22], 128, 256, 128, 256); gemm(XA, Pm[23], YF, (int)M2, 128, 256);
  bnrelu(YF, XB, M2, 256, Pm[24], Pm[25]);
  k_maxpool_h<<<(16 * 128 * 256 + 255) / 256, 256, 0, stream>>>(XB, l2f, 16 * 128, 64, 256);

  // ---- SA3: group-all, C_in=3+256=259 -> Kp=288, rows = 16*128 = 2048 ----
  k_concat3<<<(2048 * 288 + 255) / 256, 256, 0, stream>>>(l2x, l2f, XA, 2048, 256, 288);
  prepw(Pm[26], 259, 256, 288, 256);  gemm(XA, Pm[27], YF, 2048, 288, 256);
  bnrelu(YF, XB, 2048, 256, Pm[28], Pm[29]);
  prepw(Pm[30], 256, 512, 256, 512);  gemm(XB, Pm[31], YF, 2048, 256, 512);
  bnrelu(YF, XA, 2048, 512, Pm[32], Pm[33]);
  prepw(Pm[34], 512, 1024, 512, 1024); gemm(XA, Pm[35], YF, 2048, 512, 1024);
  bnrelu(YF, XB, 2048, 1024, Pm[36], Pm[37]);
  k_maxpool_h<<<(16 * 1024 + 255) / 256, 256, 0, stream>>>(XB, l3, 16, 128, 1024);

  // ---- Head (M = 16 = exactly one WMMA tile row) ----
  prepw(Pm[38], 1024, 512, 1024, 512); gemm(l3, Pm[39], YF, 16, 1024, 512);
  bnrelu(YF, h1, 16, 512, Pm[40], Pm[41]);
  prepw(Pm[42], 512, 256, 512, 256);   gemm(h1, Pm[43], YF, 16, 512, 256);
  bnrelu(YF, h2, 16, 256, Pm[44], Pm[45]);
  prepw(Pm[46], 256, 20, 256, 32);
  k_pad_vec<<<1, 64, 0, stream>>>(Pm[47], bpad, 20, 32);
  gemm(h2, bpad, YF, 16, 256, 32);
  k_copy_out<<<(16 * 20 + 255) / 256, 256, 0, stream>>>(YF, (float*)d_out, 16, 20, 32);
}
